// DistanceEncoder_61916248539247
// MI455X (gfx1250) — matrix operand order repaired and checked by reading, not compiled, not run on hardware
//
#include <hip/hip_runtime.h>
#include <hip/hip_bf16.h>

typedef _Float16 half_t;
typedef __attribute__((ext_vector_type(16))) _Float16 v16h;
typedef __attribute__((ext_vector_type(8)))  float    v8f;

#define BATCH 2
#define NPTS  512
#define HID   128
#define LDH   136        // padded LDS row stride in halves (keeps 16B alignment)
#define KSTEPS 4         // 128 / 32
#define JC    8          // j-chunks per (b, i-tile)
#define JB    (NPTS / JC)
#define NSLICES (JC * 2) // two wave-groups -> two partial slices per chunk
#define THREADS 512      // 16 waves: 8 N-tiles x 2 M-tiles (j0 / j1 rows)

union H16 { v16h v; float4 q[2]; };

// fast SiLU: a * sigmoid(a) with hardware v_rcp_f32 (no IEEE div chain)
__device__ __forceinline__ float silu(float a) {
    return a * __builtin_amdgcn_rcpf(1.0f + __expf(-a));
}

// ---- B-fragment: weights W stored row-major [K][N] (K-major). Lane l holds
// column n = ncol + (l&15); K values kbase + (l>=16 ? 8 : 0) + {0..7, 16..23},
// two halves per VGPR. Gathered from global fp32, converted to fp16. ----
__device__ __forceinline__ v16h load_bfrag(const float* __restrict__ W, int kbase, int ncol) {
    int lane = threadIdx.x & 31;
    int n  = ncol + (lane & 15);
    int k0 = kbase + ((lane & 16) ? 8 : 0);
    v16h b;
#pragma unroll
    for (int t = 0; t < 16; ++t)
        b[t] = (_Float16)W[(k0 + (t < 8 ? t : t + 8)) * HID + n];
    return b;
}

// ---- A-fragment from LDS fp16 tile (16 rows x 128 cols, stride LDH).
// Lane l: row m = l&15; K offsets {ko..ko+7, ko+16..ko+23}.
__device__ __forceinline__ v16h load_afrag(const half_t* __restrict__ hb, int kstep) {
    int lane = threadIdx.x & 31;
    int m  = lane & 15;
    int ko = kstep * 32 + ((lane & 16) ? 8 : 0);
    H16 a;
    a.q[0] = *(const float4*)(hb + m * LDH + ko);       // ds_load_b128
    a.q[1] = *(const float4*)(hb + m * LDH + ko + 16);  // ds_load_b128
    return a.v;
}

// ---- 16x128 @ 128x16 tile: 4 chained WMMAs (latency hidden by 4 waves/SIMD)
__device__ __forceinline__ v8f mm128(const half_t* __restrict__ hb, const v16h* bw) {
    v8f c = {};
#pragma unroll
    for (int k = 0; k < KSTEPS; ++k) {
        v16h a = load_afrag(hb, k);
        c = __builtin_amdgcn_wmma_f32_16x16x32_f16(false, a, false, bw[k],
                                                   (short)0, c, false, false);
    }
    return c;
}

__device__ __forceinline__ v8f bias_silu(v8f c, float bv) {
    v8f r;
#pragma unroll
    for (int v = 0; v < 8; ++v)
        r[v] = silu(c[v] + bv);
    return r;
}

// C-fragment (f32) -> fp16 LDS tile: lane l covers col ncol+(l&15), rows mb..mb+7
__device__ __forceinline__ void store_tile_h(half_t* __restrict__ hb, int ncol, v8f c) {
    int lane = threadIdx.x & 31;
    int nl = lane & 15;
    int mb = (lane & 16) ? 8 : 0;
#pragma unroll
    for (int v = 0; v < 8; ++v)
        hb[(mb + v) * LDH + ncol + nl] = (half_t)c[v];
}

// one layer stage on this wave's 16-row slice of the 32-row tile
__device__ __forceinline__ void stage(const half_t* __restrict__ src, half_t* __restrict__ dst,
                                      const v16h* bw, const float* __restrict__ bias, int ncol) {
    v8f c = mm128(src, bw);
    float bv = bias[ncol + (threadIdx.x & 15)];
    store_tile_h(dst, ncol, bias_silu(c, bv));
}

// ============ Kernel 1: fused dist -> 3x MLP -> partial sum over j ============
__global__ __launch_bounds__(THREADS)
void k1_fused(const float* __restrict__ x,
              const float* __restrict__ W10, const float* __restrict__ b10,
              const float* __restrict__ W20, const float* __restrict__ b20,
              const float* __restrict__ W11, const float* __restrict__ b11,
              const float* __restrict__ W21, const float* __restrict__ b21,
              const float* __restrict__ W12, const float* __restrict__ b12,
              const float* __restrict__ W22, const float* __restrict__ b22,
              float* __restrict__ part)
{
    __shared__ half_t hA[32 * LDH];
    __shared__ half_t hB[32 * LDH];
    __shared__ float  dsh[32];
    __shared__ float2 xi_s[16];

    const int itiles = NPTS / 16;
    int bid = blockIdx.x;
    int b   = bid / (itiles * JC);
    int rem = bid % (itiles * JC);
    int it  = rem / JC;
    int jc  = rem % JC;
    int i0  = it * 16;
    int j0  = jc * JB;

    int tid  = threadIdx.x;
    int wave = tid >> 5;          // 0..15
    int grp  = wave >> 3;         // 0: j0-rows (0-15), 1: j1-rows (16-31)
    int ncol = (wave & 7) * 16;   // feature tile
    int moff = grp * 16 * LDH;    // row offset into 32-row LDS tiles

    // Register-resident weight fragments: 5 matrices x 4 K-steps x 8 VGPRs = 160 VGPRs
    v16h w20[KSTEPS], w11[KSTEPS], w21[KSTEPS], w12[KSTEPS], w22[KSTEPS];
#pragma unroll
    for (int k = 0; k < KSTEPS; ++k) {
        w20[k] = load_bfrag(W20, k * 32, ncol);
        w11[k] = load_bfrag(W11, k * 32, ncol);
        w21[k] = load_bfrag(W21, k * 32, ncol);
        w12[k] = load_bfrag(W12, k * 32, ncol);
        w22[k] = load_bfrag(W22, k * 32, ncol);
    }
    if (tid < 16)
        xi_s[tid] = ((const float2*)x)[(size_t)b * NPTS + i0 + tid];

    v8f acc = {};   // running sum over this group's j-subset (the "mean" reduction)
    __syncthreads();

    for (int jj = 0; jj < JB; jj += 2) {      // two j's per iteration (M = 32 rows)
        if (tid < 32) {
            int r = tid & 15;
            int j = j0 + jj + (tid >> 4);
            float2 xj = ((const float2*)x)[(size_t)b * NPTS + j];
            float dx = xi_s[r].x - xj.x;
            float dy = xi_s[r].y - xj.y;
            float sq = dx * dx + dy * dy;
            dsh[tid] = sq > 0.0f ? sqrtf(sq) : 0.0f;
        }
        __syncthreads();

        // Layer 0, first linear (K=1) + SiLU, elementwise into hA (32 x 128)
#pragma unroll
        for (int s = 0; s < 8; ++s) {
            int idx = tid + THREADS * s;      // 0..4095 covers 32x128
            int rr = idx >> 7, f = idx & 127;
            float a = dsh[rr] * W10[f] + b10[f];
            hA[rr * LDH + f] = (half_t)silu(a);
        }
        __syncthreads();

        stage(hA + moff, hB + moff, w20, b20, ncol); __syncthreads();
        stage(hB + moff, hA + moff, w11, b11, ncol); __syncthreads();
        stage(hA + moff, hB + moff, w21, b21, ncol); __syncthreads();
        stage(hB + moff, hA + moff, w12, b12, ncol); __syncthreads();

        // final layer: accumulate straight into the mean accumulator (no LDS store)
        {
            v8f c = mm128(hA + moff, w22);
            acc += bias_silu(c, b22[ncol + (tid & 15)]);
        }
        __syncthreads();                       // hA consumed; safe to rewrite next iter
    }

    // Deterministic partial sums: slice = jc*2 + grp; part[slice][b*N + i][f]
    {
        int slice = jc * 2 + grp;
        int lane = tid & 31;
        int nl = lane & 15;
        int mb = (lane & 16) ? 8 : 0;
#pragma unroll
        for (int v = 0; v < 8; ++v) {
            size_t row = (size_t)b * NPTS + i0 + mb + v;
            part[((size_t)slice * BATCH * NPTS + row) * HID + ncol + nl] = acc[v];
        }
    }
}

// ============ Kernel 2: reduce partials, scale 1/N, @Wo + bo (WMMA) ============
__global__ __launch_bounds__(256)
void k2_out(const float* __restrict__ part,
            const float* __restrict__ Wo, const float* __restrict__ bo,
            float* __restrict__ out)
{
    __shared__ half_t hA[16 * LDH];
    int row0 = blockIdx.x * 16;        // global row in [0, B*N)
    int tid  = threadIdx.x;
    int wave = tid >> 5;
    int ncol = wave * 16;

    v16h wo[KSTEPS];
#pragma unroll
    for (int k = 0; k < KSTEPS; ++k)
        wo[k] = load_bfrag(Wo, k * 32, ncol);

    const float inv = 1.0f / (float)NPTS;
#pragma unroll
    for (int s = 0; s < 8; ++s) {
        int idx = tid + 256 * s;
        int rr = idx >> 7, f = idx & 127;
        size_t base = (size_t)(row0 + rr) * HID + f;
        float sum = 0.0f;
#pragma unroll
        for (int p = 0; p < NSLICES; ++p)
            sum += part[(size_t)p * BATCH * NPTS * HID + base];
        hA[rr * LDH + f] = (half_t)(sum * inv);
    }
    __syncthreads();

    v8f c = mm128(hA, wo);
    int lane = tid & 31;
    int nl = lane & 15;
    int mb = (lane & 16) ? 8 : 0;
    float bv = bo[ncol + nl];
#pragma unroll
    for (int v = 0; v < 8; ++v)
        out[(size_t)(row0 + mb + v) * HID + ncol + nl] = c[v] + bv;
}

extern "C" void kernel_launch(void* const* d_in, const int* in_sizes, int n_in,
                              void* d_out, int out_size, void* d_ws, size_t ws_size,
                              hipStream_t stream) {
    (void)in_sizes; (void)n_in; (void)out_size; (void)ws_size;
    const float* x   = (const float*)d_in[0];
    const float* W10 = (const float*)d_in[1];
    const float* b10 = (const float*)d_in[2];
    const float* W20 = (const float*)d_in[3];
    const float* b20 = (const float*)d_in[4];
    const float* W11 = (const float*)d_in[5];
    const float* b11 = (const float*)d_in[6];
    const float* W21 = (const float*)d_in[7];
    const float* b21 = (const float*)d_in[8];
    const float* W12 = (const float*)d_in[9];
    const float* b12 = (const float*)d_in[10];
    const float* W22 = (const float*)d_in[11];
    const float* b22 = (const float*)d_in[12];
    const float* Wo  = (const float*)d_in[13];
    const float* bo  = (const float*)d_in[14];

    float* part = (float*)d_ws;   // NSLICES * B * N * HID floats = 8 MB, fully overwritten

    int grid1 = BATCH * (NPTS / 16) * JC;   // 512 blocks x 512 threads (16 waves)
    k1_fused<<<grid1, dim3(THREADS), 0, stream>>>(x, W10, b10, W20, b20,
                                                  W11, b11, W21, b21,
                                                  W12, b12, W22, b22, part);
    int grid2 = BATCH * NPTS / 16;          // 64 blocks
    k2_out<<<grid2, dim3(256), 0, stream>>>(part, Wo, bo, (float*)d_out);
}
